// CrossSelfAttention_35897336660727
// MI455X (gfx1250) — matrix operand order repaired and checked by reading, not compiled, not run on hardware
//
#include <hip/hip_runtime.h>
#include <hip/hip_bf16.h>

typedef __attribute__((ext_vector_type(16))) __bf16           v16bf;
typedef __attribute__((ext_vector_type(8)))  float            v8f;
typedef __attribute__((ext_vector_type(8)))  unsigned short   v8us;

#define NPIX 4096   // H*W = 64*64
#define CCH  64
#define NB   2

// ---------------- helpers ----------------
__device__ __forceinline__ unsigned short f2bf(float x) {
    unsigned u = __float_as_uint(x);
    unsigned r = u + 0x7FFFu + ((u >> 16) & 1u);   // round-to-nearest-even
    return (unsigned short)(r >> 16);
}
__device__ __forceinline__ float bf2f(unsigned short h) {
    return __uint_as_float(((unsigned)h) << 16);
}
__device__ __forceinline__ void bf16split(float x, unsigned short& h, unsigned short& l) {
    h = f2bf(x);
    l = f2bf(x - bf2f(h));
}

// ---------------- K1: channel sums  S[mod][b][n] = sum_c x[b][c][n] ----------------
__global__ void k_chansum(const float* __restrict__ vi, const float* __restrict__ ir,
                          float* __restrict__ S) {
    int idx = blockIdx.x * blockDim.x + threadIdx.x;      // 16384 threads
    if (idx >= 2 * NB * NPIX) return;
    int n   = idx & (NPIX - 1);
    int b   = (idx >> 12) & 1;
    int mod = idx >> 13;
    const float* x = (mod ? ir : vi) + ((size_t)b * CCH) * NPIX + n;
    float s = 0.f;
    #pragma unroll 8
    for (int c = 0; c < CCH; ++c) s += x[(size_t)c * NPIX];
    S[idx] = s;
}

// ---------------- K2: sobel edge maps  E_vi, E_ir, E_q ----------------
__device__ __forceinline__ float ldS(const float* S, int h, int w) {
    return (h < 0 || h > 63 || w < 0 || w > 63) ? 0.f : S[h * 64 + w];
}
__global__ void k_sobel(const float* __restrict__ S,
                        const float* __restrict__ kxv, const float* __restrict__ kyv,
                        const float* __restrict__ kxq, const float* __restrict__ kyq,
                        const float* __restrict__ kxi, const float* __restrict__ kyi,
                        float* __restrict__ E /* E_vi | E_ir | E_q, each [2][4096] */) {
    int idx = blockIdx.x * blockDim.x + threadIdx.x;      // 8192 threads
    if (idx >= NB * NPIX) return;
    int b = idx >> 12;
    int n = idx & (NPIX - 1);
    int h = n >> 6, w = n & 63;
    const float* Sv = S + (size_t)b * NPIX;
    const float* Si = S + (size_t)(NB + b) * NPIX;
    float nv[9], ni[9];
    int i = 0;
    #pragma unroll
    for (int dh = -1; dh <= 1; ++dh)
        #pragma unroll
        for (int dw = -1; dw <= 1; ++dw) {
            nv[i] = ldS(Sv, h + dh, w + dw);
            ni[i] = ldS(Si, h + dh, w + dw);
            ++i;
        }
    float gxv = 0, gyv = 0, gxi = 0, gyi = 0, gxq = 0, gyq = 0;
    #pragma unroll
    for (int j = 0; j < 9; ++j) {
        gxv = fmaf(nv[j], kxv[j], gxv);
        gyv = fmaf(nv[j], kyv[j], gyv);
        gxi = fmaf(ni[j], kxi[j], gxi);
        gyi = fmaf(ni[j], kyi[j], gyi);
        float sq = nv[j] + ni[j];
        gxq = fmaf(sq, kxq[j], gxq);
        gyq = fmaf(sq, kyq[j], gyq);
    }
    E[idx]                   = fabsf(gxv) + fabsf(gyv);   // E_vi
    E[NB * NPIX + idx]       = fabsf(gxi) + fabsf(gyi);   // E_ir
    E[2 * NB * NPIX + idx]   = fabsf(gxq) + fabsf(gyq);   // E_q
}

// ---------------- K3: values  xv[mod][b][c][n] as bf16 hi/lo ----------------
__global__ void k_values(const float* __restrict__ vi, const float* __restrict__ ir,
                         const float* __restrict__ wv_vi, const float* __restrict__ bv_vi,
                         const float* __restrict__ wv_ir, const float* __restrict__ bv_ir,
                         unsigned short* __restrict__ xvHi, unsigned short* __restrict__ xvLo) {
    int idx = blockIdx.x * blockDim.x + threadIdx.x;      // 2*2*64*4096
    if (idx >= 2 * NB * CCH * NPIX) return;
    int n   = idx & (NPIX - 1);
    int c   = (idx >> 12) & (CCH - 1);
    int b   = (idx >> 18) & 1;
    int mod = idx >> 19;
    const float* x  = (mod ? ir : vi) + ((size_t)b * CCH) * NPIX + n;
    const float* wv = (mod ? wv_ir : wv_vi) + (size_t)c * CCH;
    float acc = (mod ? bv_ir : bv_vi)[c];
    #pragma unroll 8
    for (int k = 0; k < CCH; ++k) acc = fmaf(wv[k], x[(size_t)k * NPIX], acc);
    unsigned short h, l;
    bf16split(acc, h, l);
    xvHi[idx] = h;
    xvLo[idx] = l;
}

// ---------------- K4: alpha[m], r[m] per (attn, b) ----------------
__global__ void __launch_bounds__(256) k_alpha(const float* __restrict__ E,
                                               const float* __restrict__ w1q,
                                               const float* __restrict__ b1q,
                                               const float* __restrict__ w1vi,
                                               const float* __restrict__ w1ir,
                                               float* __restrict__ alphaA,
                                               float* __restrict__ rA) {
    int a = blockIdx.x >> 1, b = blockIdx.x & 1;          // 4 blocks
    const float* Ek = E + (size_t)a * NB * NPIX + (size_t)b * NPIX;
    const float* Eq = E + (size_t)2 * NB * NPIX + (size_t)b * NPIX;
    __shared__ float rmn[256], rmx[256];
    __shared__ float sA, sC, sMn, sMx;
    int tid = threadIdx.x;
    float mn = 3.0e38f, mx = -3.0e38f;
    for (int i = tid; i < NPIX; i += 256) {
        float v = Ek[i];
        mn = fminf(mn, v);
        mx = fmaxf(mx, v);
    }
    rmn[tid] = mn; rmx[tid] = mx;
    __syncthreads();
    for (int s = 128; s > 0; s >>= 1) {
        if (tid < s) {
            rmn[tid] = fminf(rmn[tid], rmn[tid + s]);
            rmx[tid] = fmaxf(rmx[tid], rmx[tid + s]);
        }
        __syncthreads();
    }
    if (tid == 0) {
        const float* w1k = a ? w1ir : w1vi;
        float A = 0.f, C = 0.f;
        for (int d = 0; d < 8; ++d) {
            float sq = 0.f, sk = 0.f;
            for (int c = 0; c < 2 * CCH; ++c) sq += w1q[d * 2 * CCH + c];
            for (int c = 0; c < CCH; ++c)     sk += w1k[d * CCH + c];
            A = fmaf(sq, sk, A);
            C = fmaf(b1q[d], sk, C);
        }
        sA = A; sC = C; sMn = rmn[0]; sMx = rmx[0];
    }
    __syncthreads();
    float* al = alphaA + ((size_t)(a * NB + b)) * NPIX;
    float* rr = rA     + ((size_t)(a * NB + b)) * NPIX;
    for (int m = tid; m < NPIX; m += 256) {
        float alv = fmaf(sA, Eq[m], sC);
        al[m] = alv;
        rr[m] = (alv > 0.f) ? alv * sMx : alv * sMn;
    }
}

// ---------------- K5: fused softmax x value GEMM (WMMA bf16 hi/lo) ----------------
__global__ void __launch_bounds__(256) k_attn_main(const float* __restrict__ E,
                                                   const float* __restrict__ alphaA,
                                                   const float* __restrict__ rA,
                                                   const unsigned short* __restrict__ xvHi,
                                                   const unsigned short* __restrict__ xvLo,
                                                   float* __restrict__ out) {
    __shared__ float sEk[NPIX];                            // 16 KB
    int blk = blockIdx.x;                                  // 128 blocks
    int ab  = blk >> 5;                                    // 0..3
    int grp = blk & 31;                                    // 0..31  (8 m-tiles each)
    int a = ab >> 1, b = ab & 1;

    const float* Ek = E + (size_t)a * NB * NPIX + (size_t)b * NPIX;
    for (int i = threadIdx.x; i < NPIX; i += 256) sEk[i] = Ek[i];
    __syncthreads();

    int wave = threadIdx.x >> 5;
    int lane = threadIdx.x & 31;
    int mloc = lane & 15;
    int hi16 = lane >> 4;                                  // 0 or 1
    int m0   = (grp * 8 + wave) * 16;

    size_t abBase = ((size_t)(a * NB + b)) * NPIX;
    float am = alphaA[abBase + m0 + mloc];
    float rm = rA[abBase + m0 + mloc];

    int mod = a ? 0 : 1;                                   // a=1 (ir attn) -> vi values
    const unsigned short* xh = xvHi + ((size_t)(mod * NB + b)) * CCH * NPIX;
    const unsigned short* xl = xvLo + ((size_t)(mod * NB + b)) * CCH * NPIX;

    v8f accs[4];
    #pragma unroll
    for (int t = 0; t < 4; ++t) { v8f zz = {}; accs[t] = zz; }
    float z = 0.f;
    int offA = hi16 ? 8 : 0;

    for (int k0 = 0; k0 < NPIX; k0 += 32) {
        // ---- B fragments: 32x16 probability tile (K=n, N=m), bf16 hi/lo ----
        union { unsigned short us[16]; v16bf bf; } Bh, Bl;
        int nb = k0 + hi16 * 8;
        #pragma unroll
        for (int j = 0; j < 16; ++j) {
            float e = __expf(fmaf(am, sEk[nb + j + (j & 8)], -rm));
            z += e;
            unsigned short h, l;
            bf16split(e, h, l);
            Bh.us[j] = h;
            Bl.us[j] = l;
        }
        // ---- A fragments: xv 16x32 tiles for 4 channel blocks ----
        #pragma unroll
        for (int t = 0; t < 4; ++t) {
            int c = t * 16 + mloc;
            size_t base = (size_t)c * NPIX + k0 + offA;
            union { v8us u[2]; v16bf bf; } Ah, Al;
            Ah.u[0] = *(const v8us*)(xh + base);
            Ah.u[1] = *(const v8us*)(xh + base + 16);
            Al.u[0] = *(const v8us*)(xl + base);
            Al.u[1] = *(const v8us*)(xl + base + 16);
            accs[t] = __builtin_amdgcn_wmma_f32_16x16x32_bf16(false, Ah.bf, false, Bh.bf,
                                                              (short)0, accs[t], false, false);
            accs[t] = __builtin_amdgcn_wmma_f32_16x16x32_bf16(false, Ah.bf, false, Bl.bf,
                                                              (short)0, accs[t], false, false);
            accs[t] = __builtin_amdgcn_wmma_f32_16x16x32_bf16(false, Al.bf, false, Bh.bf,
                                                              (short)0, accs[t], false, false);
        }
    }

    float zTot = z + __shfl_xor(z, 16, 32);                // fold n halves (wave32)
    float inv  = 1.0f / zTot;

    // a=1 -> vi_out (offset 0); a=0 -> ir_out (offset B*C*N)
    float* ob = out + (a ? (size_t)0 : (size_t)NB * CCH * NPIX)
                    + ((size_t)b * CCH) * NPIX;
    int mcol = m0 + mloc;
    #pragma unroll
    for (int t = 0; t < 4; ++t) {
        #pragma unroll
        for (int v = 0; v < 8; ++v) {
            int c = t * 16 + v + hi16 * 8;                 // C/D layout: lanes>=16 hold M=v+8
            ob[(size_t)c * NPIX + mcol] = accs[t][v] * inv;
        }
    }
}

// ---------------- launcher ----------------
extern "C" void kernel_launch(void* const* d_in, const int* in_sizes, int n_in,
                              void* d_out, int out_size, void* d_ws, size_t ws_size,
                              hipStream_t stream) {
    (void)in_sizes; (void)n_in; (void)out_size; (void)ws_size;
    const float* vi     = (const float*)d_in[0];
    const float* ir     = (const float*)d_in[1];
    const float* wsx_vi = (const float*)d_in[2];
    const float* wsy_vi = (const float*)d_in[3];
    const float* w1_vi  = (const float*)d_in[4];
    const float* wsx_q  = (const float*)d_in[6];
    const float* wsy_q  = (const float*)d_in[7];
    const float* w1_q   = (const float*)d_in[8];
    const float* b1_q   = (const float*)d_in[9];
    const float* wsx_ir = (const float*)d_in[10];
    const float* wsy_ir = (const float*)d_in[11];
    const float* w1_ir  = (const float*)d_in[12];
    const float* wv_vi  = (const float*)d_in[14];
    const float* bv_vi  = (const float*)d_in[15];
    const float* wv_ir  = (const float*)d_in[16];
    const float* bv_ir  = (const float*)d_in[17];
    float* out = (float*)d_out;

    // workspace layout (floats)
    float* wsf   = (float*)d_ws;
    float* S     = wsf;                        // [2 mod][2 b][4096]        = 16384
    float* E     = wsf + 16384;                // E_vi|E_ir|E_q [2][4096]*3 = 24576
    float* alpha = wsf + 40960;                // [2 a][2 b][4096]          = 16384
    float* rArr  = wsf + 57344;                // [2 a][2 b][4096]          = 16384
    unsigned short* xvHi = (unsigned short*)(wsf + 73728);            // 1,048,576 us
    unsigned short* xvLo = (unsigned short*)(wsf + 73728 + 524288);   // 1,048,576 us

    k_chansum<<<(2 * NB * NPIX) / 256, 256, 0, stream>>>(vi, ir, S);
    k_sobel<<<(NB * NPIX) / 256, 256, 0, stream>>>(S, wsx_vi, wsy_vi, wsx_q, wsy_q,
                                                   wsx_ir, wsy_ir, E);
    k_values<<<(2 * NB * CCH * NPIX) / 256, 256, 0, stream>>>(vi, ir, wv_vi, bv_vi,
                                                              wv_ir, bv_ir, xvHi, xvLo);
    k_alpha<<<4, 256, 0, stream>>>(E, w1_q, b1_q, w1_vi, w1_ir, alpha, rArr);
    k_attn_main<<<128, 256, 0, stream>>>(E, alpha, rArr, xvHi, xvLo, out);
}
